// T5LayerSelfAttention_89953795047772
// MI455X (gfx1250) — matrix operand order repaired and checked by reading, not compiled, not run on hardware
//
#include <hip/hip_runtime.h>

// ---------------------------------------------------------------------------
// T5 self-attention layer for MI455X (gfx1250, wave32, WMMA 16x16x32 f16).
// Compute-bound (~69 GFLOP vs ~60MB HBM traffic) -> everything through
// v_wmma_f32_16x16x32_f16 with f32 accumulation. K-tile staging in the
// attention loop uses the CDNA5 Tensor Data Mover (tensor_load_to_lds +
// s_wait_tensorcnt), next tiles are prefetched into L2.
// ---------------------------------------------------------------------------

typedef _Float16 f16;
typedef _Float16 v16h __attribute__((ext_vector_type(16)));
typedef _Float16 h8   __attribute__((ext_vector_type(8)));
typedef float    v8f  __attribute__((ext_vector_type(8)));
typedef unsigned int u32x4 __attribute__((ext_vector_type(4)));
typedef unsigned int u32x8 __attribute__((ext_vector_type(8)));

#define Bb   2
#define Ss   2048
#define Dd   1024
#define Hh   16
#define DKk  64
#define HDK  1024
#define BS   (Bb * Ss)       // 4096
#define EPSf 1e-6f

// ---------------- fragment helpers (ISA 7.12.2 layouts) --------------------
// A (16x32 f16): lane&15 = row M; lane>>4 = half. elems 0..7  -> K = half*8+i
//                                                elems 8..15 -> K = 16+half*8+i
__device__ __forceinline__ v16h make_a(const f16* rowp, int half) {
  h8 lo = *(const h8*)(rowp + half * 8);
  h8 hi = *(const h8*)(rowp + 16 + half * 8);
  v16h a;
#pragma unroll
  for (int i = 0; i < 8; ++i) { a[i] = lo[i]; a[i + 8] = hi[i]; }
  return a;
}
// B (32x16 f16): lane&15 = col N; elems 0..7 -> K = half*8+i, 8..15 -> K = 16+half*8+i
__device__ __forceinline__ v16h make_b(const f16* colp, int half) {
  h8 lo = *(const h8*)(colp + half * 8);
  h8 hi = *(const h8*)(colp + 16 + half * 8);
  v16h b;
#pragma unroll
  for (int i = 0; i < 8; ++i) { b[i] = lo[i]; b[i + 8] = hi[i]; }
  return b;
}
__device__ __forceinline__ v8f wmma_f16(v16h a, v16h b, v8f c) {
  return __builtin_amdgcn_wmma_f32_16x16x32_f16(false, a, false, b, (short)0, c,
                                                false, false);
}

// --------------------------- 1) RMSNorm + cast -----------------------------
__global__ __launch_bounds__(256) void rmsnorm_cast_kernel(
    const float* __restrict__ hs, const float* __restrict__ lnw,
    f16* __restrict__ xn) {
  int row = blockIdx.x;
  int tid = threadIdx.x;
  const float* x = hs + (size_t)row * Dd;
  float v[4], ss = 0.f;
#pragma unroll
  for (int i = 0; i < 4; ++i) { v[i] = x[tid + i * 256]; ss += v[i] * v[i]; }
#pragma unroll
  for (int off = 16; off >= 1; off >>= 1) ss += __shfl_xor(ss, off, 32);
  __shared__ float red[8];
  int wave = tid >> 5, lane = tid & 31;
  if (lane == 0) red[wave] = ss;
  __syncthreads();
  if (wave == 0) {
    float t = (lane < 8) ? red[lane] : 0.f;
#pragma unroll
    for (int off = 4; off >= 1; off >>= 1) t += __shfl_xor(t, off, 32);
    if (lane == 0) red[0] = t;
  }
  __syncthreads();
  float inv = rsqrtf(red[0] / (float)Dd + EPSf);
#pragma unroll
  for (int i = 0; i < 4; ++i) {
    int c = tid + i * 256;
    xn[(size_t)row * Dd + c] = (f16)(v[i] * inv * lnw[c]);
  }
}

// ------------------- 2) weight transpose + cast to f16 ---------------------
__global__ __launch_bounds__(256) void transpose_cast_kernel(
    const float* __restrict__ src, f16* __restrict__ dst) {
  int idx = blockIdx.x * 256 + threadIdx.x;
  int r = idx >> 10, c = idx & 1023;
  dst[(size_t)c * 1024 + r] = (f16)src[idx];
}

// ------------------- 3) T5 relative-position bias table --------------------
// btab[h][rel + 2048], rel = k - q.
__global__ __launch_bounds__(256) void bias_table_kernel(
    const float* __restrict__ rel_emb, float* __restrict__ btab) {
  int idx = blockIdx.x * 256 + threadIdx.x;  // h*4096 + i
  int h = idx >> 12, i = idx & 4095;
  int rel = i - 2048;
  int bucket = rel > 0 ? 16 : 0;
  int rp = rel < 0 ? -rel : rel;
  int val;
  if (rp < 8) {
    val = rp;
  } else {
    int vl = 8 + (int)(logf((float)rp * 0.125f) / logf(16.0f) * 8.0f);
    val = vl < 15 ? vl : 15;
  }
  bucket += val;
  btab[idx] = rel_emb[bucket * Hh + h];
}

// ---------------- 4) QKV projection GEMM (f16 in/out, f32 acc) -------------
__global__ __launch_bounds__(256) void gemm_qkv_kernel(
    const f16* __restrict__ A, const f16* __restrict__ Bt,
    f16* __restrict__ out) {
  int tid = threadIdx.x;
  int wave = tid >> 5, lane = tid & 31;
  int half = lane >> 4, n = lane & 15;
  int bm = blockIdx.x >> 4;     // 128 row-blocks of 32
  int bn = blockIdx.x & 15;     // 16 col-blocks of 64
  int row0 = (bm * 2 + (wave >> 2)) * 16;
  int col0 = (bn * 4 + (wave & 3)) * 16;
  const f16* arow = A + (size_t)(row0 + n) * Dd;
  const f16* bcol = Bt + (size_t)(col0 + n) * Dd;
  v8f acc = {};
  for (int k0 = 0; k0 < Dd; k0 += 32)
    acc = wmma_f16(make_a(arow + k0, half), make_b(bcol + k0, half), acc);
  int c = col0 + n;
  int h = c >> 6, dk = c & 63;
#pragma unroll
  for (int r = 0; r < 8; ++r) {
    int R = row0 + half * 8 + r;
    int b = R >> 11, s = R & 2047;
    out[((size_t)(b * Hh + h) * Ss + s) * DKk + dk] = (f16)acc[r];
  }
}

// -------------------- 5) flash attention with T5 bias ----------------------
// 4 waves/block, each owns one 16-row Q tile. K tile staged by the Tensor
// Data Mover (tensor_load_to_lds, TENSORcnt); V tile transposed via VALU/DS.
__global__ __launch_bounds__(128) void attn_kernel(
    const f16* __restrict__ Q, const f16* __restrict__ K,
    const f16* __restrict__ V, const float* __restrict__ btab,
    f16* __restrict__ ctx) {
  __shared__ f16 kbuf[32][64];      // [key][dk]  (TDM destination)
  __shared__ f16 vbufT[64][32];     // [dk][key]
  __shared__ f16 pbuf[4][16][40];   // per-wave P transpose buffer

  int tid = threadIdx.x;
  int wave = tid >> 5, lane = tid & 31;
  int half = lane >> 4, n = lane & 15;
  int bh = blockIdx.x >> 5;         // b*H + h
  int qg = blockIdx.x & 31;
  int h = bh & 15;
  int qbase = (qg * 4 + wave) * 16;

  const f16* qrow = Q + ((size_t)bh * Ss + qbase + n) * DKk;
  v16h a0 = make_a(qrow, half);
  v16h a1 = make_a(qrow + 32, half);

  v8f zero = {};
  v8f acc[4] = {zero, zero, zero, zero};
  float mrow[8], lrow[8];
#pragma unroll
  for (int r = 0; r < 8; ++r) { mrow[r] = -1e30f; lrow[r] = 0.f; }

  const f16* kbase = K + (size_t)bh * Ss * DKk;
  const f16* vbase = V + (size_t)bh * Ss * DKk;
  int srow = tid >> 2;              // staging: key row 0..31
  int sch = tid & 3;                // 16-half chunk 0..3

  // Constant parts of the TDM descriptor (D# group 0/1, ISA 08 §8.3/8.4):
  // 2D tile: 64 elem/row (2B) x 32 rows, tensor == tile, dim0_stride = 64.
  unsigned lds_off = (unsigned)(size_t)(&kbuf[0][0]);
  u32x8 g1;
  g1[0] = (1u << 16);        // wg_mask=0, data_size=1 (2 bytes)
  g1[1] = (64u << 16);       // tensor_dim0 = 64 (bits[79:48] low half)
  g1[2] = (32u << 16);       // tensor_dim0 hi=0 | tensor_dim1 = 32 (low half)
  g1[3] = (64u << 16);       // tensor_dim1 hi=0 | tile_dim0 = 64
  g1[4] = 32u;               // tile_dim1 = 32, tile_dim2 = 0
  g1[5] = 64u;               // tensor_dim0_stride = 64
  g1[6] = 0u;                // stride hi / tensor_dim1_stride lo
  g1[7] = 0u;

  for (int kb = 0; kb < Ss; kb += 32) {
    __syncthreads();
    if (wave == 0) {
      // D# group 0: count=1 | lds_addr | global_addr[56:0] | type=2
      unsigned long long ga =
          (unsigned long long)(size_t)(kbase + (size_t)kb * DKk);
      u32x4 g0;
      g0[0] = 1u;
      g0[1] = lds_off;
      g0[2] = (unsigned)(ga & 0xffffffffu);
      g0[3] = (unsigned)((ga >> 32) & 0x01ffffffu) | (2u << 30);
      asm volatile("tensor_load_to_lds %0, %1" ::"s"(g0), "s"(g1) : "memory");
    }
    {  // stage transposed V tile (VALU path) + prefetch next K/V tiles to L2
      const f16* vg = vbase + (size_t)(kb + srow) * DKk + sch * 16;
      h8 v0 = *(const h8*)vg, v1 = *(const h8*)(vg + 8);
#pragma unroll
      for (int i = 0; i < 8; ++i) {
        vbufT[sch * 16 + i][srow] = v0[i];
        vbufT[sch * 16 + 8 + i][srow] = v1[i];
      }
      if (kb + 32 < Ss) {
        __builtin_prefetch(vg + 32 * DKk);
        __builtin_prefetch(kbase + (size_t)(kb + 32 + srow) * DKk + sch * 16);
      }
    }
    if (wave == 0)
      asm volatile("s_wait_tensorcnt 0x0" ::: "memory");
    __syncthreads();

    // scores: two 16x16 tiles (keys kb..+15, kb+16..+31), K-dim = DK = 64
    v8f s0 = {}, s1 = {};
    s0 = wmma_f16(a0, make_b(&kbuf[n][0], half), s0);
    s0 = wmma_f16(a1, make_b(&kbuf[n][32], half), s0);
    s1 = wmma_f16(a0, make_b(&kbuf[16 + n][0], half), s1);
    s1 = wmma_f16(a1, make_b(&kbuf[16 + n][32], half), s1);

    // add relative-position bias (table cached in L2)
    const float* bp = btab + h * 4096 + 2048 + kb + n - (qbase + half * 8);
#pragma unroll
    for (int r = 0; r < 8; ++r) { s0[r] += bp[-r]; s1[r] += bp[16 - r]; }

    // online softmax: rows live in 16-lane halves -> xor-butterfly over 8..1
    float mnew[8], scale[8];
#pragma unroll
    for (int r = 0; r < 8; ++r) {
      float v = fmaxf(s0[r], s1[r]);
#pragma unroll
      for (int off = 8; off >= 1; off >>= 1)
        v = fmaxf(v, __shfl_xor(v, off, 32));
      mnew[r] = fmaxf(mrow[r], v);
      scale[r] = __expf(mrow[r] - mnew[r]);
      mrow[r] = mnew[r];
    }
#pragma unroll
    for (int r = 0; r < 8; ++r) {
      float p0 = __expf(s0[r] - mnew[r]);
      float p1 = __expf(s1[r] - mnew[r]);
      float rs = p0 + p1;
#pragma unroll
      for (int off = 8; off >= 1; off >>= 1) rs += __shfl_xor(rs, off, 32);
      lrow[r] = lrow[r] * scale[r] + rs;
      pbuf[wave][half * 8 + r][n] = (f16)p0;
      pbuf[wave][half * 8 + r][16 + n] = (f16)p1;
#pragma unroll
      for (int t = 0; t < 4; ++t) acc[t][r] *= scale[r];
    }

    // P back out of LDS as an A fragment (lane = row)
    v16h ap;
    {
      h8 lo = *(const h8*)&pbuf[wave][n][half * 8];
      h8 hi = *(const h8*)&pbuf[wave][n][16 + half * 8];
#pragma unroll
      for (int i = 0; i < 8; ++i) { ap[i] = lo[i]; ap[i + 8] = hi[i]; }
    }
    // ctx += P(16x32) @ V(32x64): 4 column tiles
#pragma unroll
    for (int t = 0; t < 4; ++t)
      acc[t] = wmma_f16(ap, make_b(&vbufT[t * 16 + n][0], half), acc[t]);
  }

  // epilogue: normalize and store ctx as [B,S,H*DK] f16
  int b = bh >> 4;
#pragma unroll
  for (int t = 0; t < 4; ++t)
#pragma unroll
    for (int r = 0; r < 8; ++r) {
      int qr = qbase + half * 8 + r;
      ctx[((size_t)b * Ss + qr) * HDK + h * 64 + t * 16 + n] =
          (f16)(acc[t][r] / lrow[r]);
    }
}

// ------------------ 6) output projection + residual (f32) ------------------
__global__ __launch_bounds__(256) void gemm_out_kernel(
    const f16* __restrict__ A, const f16* __restrict__ Bt,
    const float* __restrict__ hs, float* __restrict__ out) {
  int tid = threadIdx.x;
  int wave = tid >> 5, lane = tid & 31;
  int half = lane >> 4, n = lane & 15;
  int bm = blockIdx.x >> 4;
  int bn = blockIdx.x & 15;
  int row0 = (bm * 2 + (wave >> 2)) * 16;
  int col0 = (bn * 4 + (wave & 3)) * 16;
  const f16* arow = A + (size_t)(row0 + n) * HDK;
  const f16* bcol = Bt + (size_t)(col0 + n) * HDK;
  v8f acc = {};
  for (int k0 = 0; k0 < HDK; k0 += 32)
    acc = wmma_f16(make_a(arow + k0, half), make_b(bcol + k0, half), acc);
  int c = col0 + n;
#pragma unroll
  for (int r = 0; r < 8; ++r) {
    size_t R = row0 + half * 8 + r;
    out[R * Dd + c] = acc[r] + hs[R * Dd + c];
  }
}

// ---------------------------------------------------------------------------
extern "C" void kernel_launch(void* const* d_in, const int* in_sizes, int n_in,
                              void* d_out, int out_size, void* d_ws,
                              size_t ws_size, hipStream_t stream) {
  (void)in_sizes; (void)n_in; (void)out_size; (void)ws_size;
  const float* hs  = (const float*)d_in[0];
  const float* lnw = (const float*)d_in[1];
  const float* wq  = (const float*)d_in[2];
  const float* wk  = (const float*)d_in[3];
  const float* wv  = (const float*)d_in[4];
  const float* wo  = (const float*)d_in[5];
  const float* rel = (const float*)d_in[6];
  float* out = (float*)d_out;

  char* ws = (char*)d_ws;
  constexpr size_t XN_B  = (size_t)BS * Dd * 2;             // 8 MiB
  constexpr size_t WT_B  = (size_t)Dd * HDK * 2;            // 2 MiB
  constexpr size_t QKV_B = (size_t)Bb * Hh * Ss * DKk * 2;  // 8 MiB
  size_t o = 0;
  f16* xn   = (f16*)(ws + o);            // also reused as ctx after QKV GEMMs
  o += XN_B;
  f16* wq_t = (f16*)(ws + o); o += WT_B;
  f16* wk_t = (f16*)(ws + o); o += WT_B;
  f16* wv_t = (f16*)(ws + o); o += WT_B;
  f16* wo_t = (f16*)(ws + o); o += WT_B;
  f16* qf   = (f16*)(ws + o); o += QKV_B;
  f16* kf   = (f16*)(ws + o); o += QKV_B;
  f16* vf   = (f16*)(ws + o); o += QKV_B;
  float* btab = (float*)(ws + o);        // H * 4096 f32
  f16* ctx = xn;                         // xn dead after projections

  rmsnorm_cast_kernel<<<BS, 256, 0, stream>>>(hs, lnw, xn);
  transpose_cast_kernel<<<4096, 256, 0, stream>>>(wq, wq_t);
  transpose_cast_kernel<<<4096, 256, 0, stream>>>(wk, wk_t);
  transpose_cast_kernel<<<4096, 256, 0, stream>>>(wv, wv_t);
  transpose_cast_kernel<<<4096, 256, 0, stream>>>(wo, wo_t);
  bias_table_kernel<<<(Hh * 4096) / 256, 256, 0, stream>>>(rel, btab);
  gemm_qkv_kernel<<<2048, 256, 0, stream>>>(xn, wq_t, qf);
  gemm_qkv_kernel<<<2048, 256, 0, stream>>>(xn, wk_t, kf);
  gemm_qkv_kernel<<<2048, 256, 0, stream>>>(xn, wv_t, vf);
  attn_kernel<<<1024, 128, 0, stream>>>(qf, kf, vf, btab, ctx);
  gemm_out_kernel<<<2048, 256, 0, stream>>>(ctx, wo_t, hs, out);
}
